// ScaledDotProductAttention_20985210208822
// MI455X (gfx1250) — matrix operand order repaired and checked by reading, not compiled, not run on hardware
//
#include <hip/hip_runtime.h>
#include <hip/hip_bf16.h>

typedef __attribute__((ext_vector_type(16))) _Float16 v16h;
typedef __attribute__((ext_vector_type(8)))  float    v8f;

#define BB 32
#define LL 2048
#define DD 64
#define SCALE 0.125f

// xor-butterfly within 16-lane rows via DPP16 ROW_XMASK (pure VALU, no LDS pipe).
// dpp_ctrl = 0x160 | xor_mask ; row_mask/bank_mask = 0xf ; bound_ctrl = true.
template <int CTRL>
__device__ __forceinline__ float dpp_xor(float x) {
  int y = __builtin_amdgcn_update_dpp(0, __float_as_int(x), CTRL, 0xf, 0xf, true);
  return __int_as_float(y);
}
__device__ __forceinline__ float row_max16(float x) {
  x = fmaxf(x, dpp_xor<0x161>(x));
  x = fmaxf(x, dpp_xor<0x162>(x));
  x = fmaxf(x, dpp_xor<0x164>(x));
  x = fmaxf(x, dpp_xor<0x168>(x));
  return x;
}
__device__ __forceinline__ float row_sum16(float x) {
  x += dpp_xor<0x161>(x);
  x += dpp_xor<0x162>(x);
  x += dpp_xor<0x164>(x);
  x += dpp_xor<0x168>(x);
  return x;
}

// One wave (32 lanes) owns a 16-query tile. Block = 8 waves = 128 queries.
// Pass 1: online softmax stats (running max/sum) over all keys via QK^T WMMA.
// Pass 2: recompute scores, write normalized attention once, fuse P@V WMMA.
__global__ __launch_bounds__(256)
void attn_fused_kernel(const float* __restrict__ Q, const float* __restrict__ K,
                       const float* __restrict__ V, float* __restrict__ ctx,
                       float* __restrict__ att) {
  __shared__ _Float16 lds_p[8][16][32];   // per-wave 16x32 f16 P staging (C-layout -> A-layout)

  const int lane = threadIdx.x & 31;
  const int wave = threadIdx.x >> 5;
  const int b    = blockIdx.x >> 4;                        // 16 query-blocks per batch
  const int q0   = ((blockIdx.x & 15) << 7) + (wave << 4); // 128 queries per block
  const int n    = lane & 15;                              // column index (B/C layouts)
  const int hi   = lane >> 4;                              // half-wave selector

  const float* Qb = Q + (size_t)b * LL * DD;
  const float* Kb = K + (size_t)b * LL * DD;
  const float* Vb = V + (size_t)b * LL * DD;

  // ---------- Q A-fragments (scale folded in), two 32-wide d-tiles ----------
  // A 16x32 f16 operand, contiguous k order used here (B matches): lane (m = lane&15, hi): element e -> k = hi*16+e
  v16h qf[2];
  {
    const float* qrow = Qb + (size_t)(q0 + n) * DD;
#pragma unroll
    for (int t = 0; t < 2; ++t) {
      const float* p0 = qrow + t * 32 + hi * 16;
      const float* p1 = qrow + t * 32 + hi * 16 + 8;
#pragma unroll
      for (int e = 0; e < 8; ++e) {
        qf[t][e]     = (_Float16)(p0[e] * SCALE);
        qf[t][e + 8] = (_Float16)(p1[e] * SCALE);
      }
    }
  }

  // ---------- S tile: 16 queries x 16 keys, f32 accum, K-dim = 64 (2 WMMAs) ----------
  // K B-fragment: B 32x16 f16, same contiguous k order as A: lane n holds column n; element e -> kdim = hi*16+e.
  auto qk_tile = [&](int k0) -> v8f {
    v8f c = {0.f, 0.f, 0.f, 0.f, 0.f, 0.f, 0.f, 0.f};
#pragma unroll
    for (int t = 0; t < 2; ++t) {
      v16h kf;
      const float* kp = Kb + (size_t)(k0 + n) * DD + t * 32 + hi * 16;
#pragma unroll
      for (int e = 0; e < 16; ++e) kf[e] = (_Float16)kp[e];
      c = __builtin_amdgcn_wmma_f32_16x16x32_f16(false, qf[t], false, kf,
                                                 (short)0, c, false, false);
    }
    return c;
  };

  // ---------- pass 1: online max / sum over all 2048 keys ----------
  float rm[8], rs[8];
#pragma unroll
  for (int v = 0; v < 8; ++v) { rm[v] = -3.0e38f; rs[v] = 0.0f; }

  for (int kt = 0; kt < LL / 16; ++kt) {
    v8f s = qk_tile(kt * 16);
#pragma unroll
    for (int v = 0; v < 8; ++v) {
      float tm = row_max16(s[v]);
      float nm = fmaxf(rm[v], tm);
      float ex = __expf(s[v] - nm);
      float ts = row_sum16(ex);
      rs[v] = rs[v] * __expf(rm[v] - nm) + ts;
      rm[v] = nm;
    }
  }

  float rinv[8];
#pragma unroll
  for (int v = 0; v < 8; ++v) rinv[v] = 1.0f / rs[v];

  // ---------- pass 2: recompute, emit attention, fuse P@V ----------
  v8f cacc[4];
#pragma unroll
  for (int dt = 0; dt < 4; ++dt) cacc[dt] = (v8f){0.f,0.f,0.f,0.f,0.f,0.f,0.f,0.f};

  float* attq = att + ((size_t)b * LL + q0) * LL;
  _Float16 (*pl)[32] = lds_p[wave];

  for (int kg = 0; kg < LL / 32; ++kg) {
    const int kbase = kg * 32;
#pragma unroll
    for (int h2 = 0; h2 < 2; ++h2) {
      const int k0 = kbase + h2 * 16;
      v8f s = qk_tile(k0);
#pragma unroll
      for (int v = 0; v < 8; ++v) {
        float p = __expf(s[v] - rm[v]) * rinv[v];   // exact softmax prob
        const int row = v + 8 * hi;                  // C layout: M = vgpr + 8*hi
        attq[(size_t)row * LL + k0 + n] = p;         // attention output (fp32)
        pl[row][h2 * 16 + n] = (_Float16)p;          // stage for A-layout reload
      }
    }
    // order LDS stores (all lanes) before A-fragment reload; DS ops are in-order per wave
    asm volatile("s_wait_dscnt 0" ::: "memory");

    // P A-fragment (16x32): lane m = lane&15; element e -> kk = hi*16+e
    v16h pf;
#pragma unroll
    for (int e = 0; e < 8; ++e) {
      pf[e]     = pl[n][hi * 16 + e];
      pf[e + 8] = pl[n][hi * 16 + 8 + e];
    }

    // V B-fragments: lane n holds d-column (dt*16+n); element e -> key = kbase + hi*16 + e
#pragma unroll
    for (int dt = 0; dt < 4; ++dt) {
      v16h vf;
      const float* vp = Vb + (size_t)(kbase + hi * 16) * DD + dt * 16 + n;
#pragma unroll
      for (int e = 0; e < 16; ++e) vf[e] = (_Float16)vp[(size_t)e * DD];
      cacc[dt] = __builtin_amdgcn_wmma_f32_16x16x32_f16(false, pf, false, vf,
                                                        (short)0, cacc[dt], false, false);
    }
  }

  // ---------- write context [16 x 64] ----------
  float* cb = ctx + ((size_t)b * LL + q0) * DD;
#pragma unroll
  for (int dt = 0; dt < 4; ++dt)
#pragma unroll
    for (int v = 0; v < 8; ++v)
      cb[(size_t)(v + 8 * hi) * DD + dt * 16 + n] = cacc[dt][v];
}

extern "C" void kernel_launch(void* const* d_in, const int* in_sizes, int n_in,
                              void* d_out, int out_size, void* d_ws, size_t ws_size,
                              hipStream_t stream) {
  const float* q = (const float*)d_in[0];
  const float* k = (const float*)d_in[1];
  const float* v = (const float*)d_in[2];
  float* ctx = (float*)d_out;
  float* att = ctx + (size_t)BB * LL * DD;   // outputs concatenated: (context, attention)

  dim3 grid(BB * (LL / 128));  // 512 blocks
  dim3 block(256);             // 8 waves, one 16-query tile each
  attn_fused_kernel<<<grid, block, 0, stream>>>(q, k, v, ctx, att);
}
